// GMSA_59098749993512
// MI455X (gfx1250) — compile-verified, matrix operand order checked by reading, not run on hardware
//
#include <hip/hip_runtime.h>
#include <hip/hip_bf16.h>
#include <cstdint>
#include <cstddef>

// ---------------------------------------------------------------------------
// GMSA for MI455X (gfx1250): bf16 WMMA everywhere, f32 accumulate.
// Bandwidth-bound (~200MB moved vs ~30 GFLOP) -> bf16 intermediates,
// LN fused into GEMM-1, shift-roll folded into attention addressing,
// weights pre-converted to bf16, B-operands hoisted per wave,
// vectorized LDS softmax + vectorized window gathers.
// ---------------------------------------------------------------------------

typedef __attribute__((ext_vector_type(16))) __bf16 v16bf;
typedef __attribute__((ext_vector_type(4)))  __bf16 v4bf;
typedef __attribute__((ext_vector_type(2)))  __bf16 v2bf;
typedef __attribute__((ext_vector_type(8)))  float  v8f;

#define WMMA_BF16(A, B, C) \
  __builtin_amdgcn_wmma_f32_16x16x32_bf16(false, (A), false, (B), (short)0, (C), false, false)

static constexpr int CH  = 192;
static constexpr int NB  = 2;
static constexpr int DIM = 32;
static constexpr int VOX = DIM * DIM * DIM;   // 32768 voxels per batch

static constexpr int W_IN_ELEMS  = 3 * CH * CH;   // 110592
static constexpr int W_OUT_ELEMS = CH * CH;       // 36864

// ---------------------------------------------------------------------------
// Kernel 0: one-shot f32 -> bf16 conversion of both weight matrices.
// ---------------------------------------------------------------------------
__global__ __launch_bounds__(256) void k_cvt_weights(
    const float* __restrict__ Win, const float* __restrict__ Wout,
    __bf16* __restrict__ WinB, __bf16* __restrict__ WoutB)
{
  int idx = blockIdx.x * 256 + threadIdx.x;
  if (idx < W_IN_ELEMS) WinB[idx] = (__bf16)Win[idx];
  if (idx < W_OUT_ELEMS) WoutB[idx] = (__bf16)Wout[idx];
}

// ---------------------------------------------------------------------------
// Kernel 1: LayerNorm(C) fused with 192->576 projection.
// Block = 64 voxels of one batch, 256 threads (8 waves).
// P layout: [b][o][v] bf16, o = s*192 + qkv*64 + head*32 + c.
// Each wave owns a fixed n-tile -> its 6 B-operands are loaded once.
// ---------------------------------------------------------------------------
__global__ __launch_bounds__(256) void k_ln_proj(
    const float* __restrict__ X, const float* __restrict__ G,
    const float* __restrict__ Bt, const __bf16* __restrict__ WinB,
    const float* __restrict__ Bin, __bf16* __restrict__ P)
{
  __shared__ float Xraw[CH * 64];                              // 48 KB
  __shared__ __attribute__((aligned(64))) __bf16 Xn[64 * CH];  // 24 KB, [v][c]
  __shared__ float Sm[4 * 64], Sq[4 * 64], Mu[64], Rs[64];

  const int tid  = threadIdx.x;
  const int gv0  = blockIdx.x * 64;
  const int b    = gv0 >> 15;
  const int v0   = gv0 & (VOX - 1);
  const int vloc = tid & 63, crow = tid >> 6;

  float s = 0.f, sq = 0.f;
  for (int it = 0; it < 48; ++it) {                     // coalesced channel sweep
    int c = it * 4 + crow;
    float val = X[(size_t)(b * CH + c) * VOX + v0 + vloc];
    Xraw[c * 64 + vloc] = val;
    s += val; sq += val * val;
  }
  Sm[crow * 64 + vloc] = s;
  Sq[crow * 64 + vloc] = sq;
  __syncthreads();
  if (tid < 64) {
    float ts = Sm[tid] + Sm[64 + tid] + Sm[128 + tid] + Sm[192 + tid];
    float tq = Sq[tid] + Sq[64 + tid] + Sq[128 + tid] + Sq[192 + tid];
    float mu = ts * (1.f / CH);
    float var = tq * (1.f / CH) - mu * mu;
    Mu[tid] = mu;
    Rs[tid] = rsqrtf(var + 1e-5f);
  }
  __syncthreads();
  for (int it = 0; it < 48; ++it) {
    int idx = it * 256 + tid;
    int v = idx / CH, c = idx - v * CH;
    float xn = (Xraw[c * 64 + v] - Mu[v]) * Rs[v] * G[c] + Bt[c];
    Xn[v * CH + c] = (__bf16)xn;
  }
  __syncthreads();

  const int wave = tid >> 5, lane = tid & 31, ln = lane & 15, lh = lane >> 4;
  const int n0 = (wave & 3) * 16;                       // fixed per wave

  v16bf bvs[6];                                         // hoisted B-operands
#pragma unroll
  for (int kk = 0; kk < 6; ++kk)
    bvs[kk] = *(const v16bf*)(Xn + (n0 + ln) * CH + kk * 32 + lh * 16);

  for (int tile = wave; tile < 144; tile += 8) {        // m0 varies, n0 fixed
    const int m0 = (tile >> 2) * 16;
    v8f acc = {};
#pragma unroll
    for (int kk = 0; kk < 6; ++kk) {
      v16bf a = *(const v16bf*)(WinB + (size_t)(m0 + ln) * CH + kk * 32 + lh * 16);
      acc = WMMA_BF16(a, bvs[kk], acc);
    }
#pragma unroll
    for (int r = 0; r < 8; ++r) {
      int o = m0 + r + 8 * lh;
      float val = acc[r] + Bin[o];
      P[(size_t)(b * 3 * CH + o) * VOX + v0 + n0 + ln] = (__bf16)val;
    }
  }
}

// ---------------------------------------------------------------------------
// Kernel 2: shifted-window attention, one block per (b, window-group, head).
// WS=2 packs 2 adjacent-W windows into one 16-token tile (block-diag mask is
// uniform per lane-half; K-dim zero-padded to 32). 128 threads (4 waves).
// ---------------------------------------------------------------------------
template <int WS>
__global__ __launch_bounds__(128) void k_attn(const __bf16* __restrict__ P,
                                              __bf16* __restrict__ Y)
{
  constexpr int S   = (WS == 2) ? 0 : ((WS == 4) ? 1 : 2);
  constexpr int T   = (WS == 2) ? 16 : WS * WS * WS;    // tokens per tile group
  constexpr int TK  = (WS == 2) ? 32 : T;               // padded K-dim for AV
  constexpr int SH  = WS / 2;                           // shift
  constexpr int NIT = T / 16;
  constexpr int NJT = T / 16;
  constexpr int NKT = TK / 32;
  constexpr int NG  = (WS == 2) ? 2048 : ((WS == 4) ? 512 : 64);

  __shared__ __attribute__((aligned(64))) __bf16 Qs[T * 32];
  __shared__ __attribute__((aligned(64))) __bf16 Ks[T * 32];
  __shared__ __attribute__((aligned(64))) __bf16 Vs[32 * TK];
  __shared__ __attribute__((aligned(64))) float Sw[4 * 16 * T];
  __shared__ __attribute__((aligned(64))) __bf16 Sb[4 * 16 * TK];

  const int gid  = blockIdx.x;
  const int head = gid & 1;
  const int rest = gid >> 1;
  const int g    = rest & (NG - 1);
  const int b    = rest / NG;

  int D0, H0, W0;
  if (WS == 2)      { D0 = (g >> 7) * 2; H0 = ((g >> 3) & 15) * 2; W0 = (g & 7) * 4; }
  else if (WS == 4) { D0 = (g >> 6) * 4; H0 = ((g >> 3) & 7) * 4;  W0 = (g & 7) * 4; }
  else              { D0 = (g >> 4) * 8; H0 = ((g >> 2) & 3) * 8;  W0 = (g & 3) * 8; }

  auto vox = [&](int t) -> int {
    int z, y, xw;
    if (WS == 2) {
      int win = t >> 3, l = t & 7;
      z = D0 + ((l >> 2) & 1); y = H0 + ((l >> 1) & 1); xw = W0 + win * 2 + (l & 1);
    } else {
      z = D0 + t / (WS * WS); y = H0 + (t / WS) % WS; xw = W0 + t % WS;
    }
    return (((z + SH) & 31) << 10) | (((y + SH) & 31) << 5) | ((xw + SH) & 31);
  };

  const __bf16* qB = P + ((size_t)b * 3 * CH + S * CH + 0   + head * 32) * VOX;
  const __bf16* kB = P + ((size_t)b * 3 * CH + S * CH + 64  + head * 32) * VOX;
  const __bf16* vB = P + ((size_t)b * 3 * CH + S * CH + 128 + head * 32) * VOX;

  const int tid = threadIdx.x;
  if (WS == 2) {                                        // tiny branch: scalar gather
    for (int idx = tid; idx < 32 * T; idx += 128) {
      int c = idx / T, t = idx - c * T;
      size_t off = (size_t)c * VOX + vox(t);
      Qs[t * 32 + c] = qB[off];
      Ks[t * 32 + c] = kB[off];
      Vs[c * TK + t] = vB[off];
    }
    for (int idx = tid; idx < 32 * 16; idx += 128) {    // zero-pad V tokens 16..31
      int c = idx >> 4, t = 16 + (idx & 15);
      Vs[c * TK + t] = (__bf16)0.f;
    }
  } else {
    // vectorized gather: 4 tokens along W per group; wrap only if x0 > 28
    constexpr int GPC = T / 4;
    for (int gi = tid; gi < 32 * GPC; gi += 128) {
      int c = gi / GPC;
      int tg = (gi - c * GPC) * 4;
      int z  = D0 + tg / (WS * WS);
      int y  = H0 + (tg / WS) % WS;
      int x0 = (W0 + (tg % WS) + SH) & 31;              // even (W0,SH,tg%WS even)
      size_t cb = (size_t)c * VOX + (((z + SH) & 31) << 10) + (((y + SH) & 31) << 5);
      if (x0 <= 28) {                                   // contiguous 4-token run
        const uint32_t* qp = (const uint32_t*)(qB + cb + x0);
        const uint32_t* kp = (const uint32_t*)(kB + cb + x0);
        const uint32_t* vp = (const uint32_t*)(vB + cb + x0);
        union { uint32_t u; v2bf h; } q0{qp[0]}, q1{qp[1]}, k0{kp[0]}, k1{kp[1]};
        uint32_t w0v = vp[0], w1v = vp[1];
        Qs[(tg + 0) * 32 + c] = q0.h[0];
        Qs[(tg + 1) * 32 + c] = q0.h[1];
        Qs[(tg + 2) * 32 + c] = q1.h[0];
        Qs[(tg + 3) * 32 + c] = q1.h[1];
        Ks[(tg + 0) * 32 + c] = k0.h[0];
        Ks[(tg + 1) * 32 + c] = k0.h[1];
        Ks[(tg + 2) * 32 + c] = k1.h[0];
        Ks[(tg + 3) * 32 + c] = k1.h[1];
        *(uint32_t*)&Vs[c * TK + tg]     = w0v;
        *(uint32_t*)&Vs[c * TK + tg + 2] = w1v;
      } else {                                          // wrapping group (rare)
#pragma unroll
        for (int j = 0; j < 4; ++j) {
          size_t off = cb + ((x0 + j) & 31);
          Qs[(tg + j) * 32 + c] = qB[off];
          Ks[(tg + j) * 32 + c] = kB[off];
          Vs[c * TK + tg + j]   = vB[off];
        }
      }
    }
  }
  __syncthreads();

  const int wave = tid >> 5, lane = tid & 31, ln = lane & 15, lh = lane >> 4;
  float*  Swr = Sw + wave * 16 * T;
  __bf16* Sbr = Sb + wave * 16 * TK;

  for (int it = wave; it < NIT; it += 4) {
    v16bf aq = *(const v16bf*)(Qs + (it * 16 + ln) * 32 + lh * 16);
    for (int jt = 0; jt < NJT; ++jt) {                  // S = Q K^T  (K-dim = 32 = c)
      v16bf bk = *(const v16bf*)(Ks + (jt * 16 + ln) * 32 + lh * 16);
      v8f sacc = {};
      sacc = WMMA_BF16(aq, bk, sacc);
#pragma unroll
      for (int r = 0; r < 8; ++r)
        Swr[(r + 8 * lh) * T + jt * 16 + ln] = sacc[r];
    }
    // row softmax, vectorized: lane pair (ln, ln+16) splits the row.
    // For WS==2 the block-diagonal mask is uniform per half: lh == row>>3.
    {
      const int row = ln;
      const bool vh = (WS != 2) || (lh == (row >> 3));
      const int c0 = lh * (T / 2);
      float* Sr = Swr + row * T;
      float m = -1e30f;
      if (vh) {
        for (int c = c0; c < c0 + T / 2; c += 4) {
          float4 sv = *(const float4*)(Sr + c);
          m = fmaxf(m, fmaxf(fmaxf(sv.x, sv.y), fmaxf(sv.z, sv.w)));
        }
      }
      m = fmaxf(m, __shfl_xor(m, 16, 32));
      float sum = 0.f;
      if (vh) {
        for (int c = c0; c < c0 + T / 2; c += 4) {
          float4 sv = *(const float4*)(Sr + c);
          sv.x = __expf(sv.x - m); sv.y = __expf(sv.y - m);
          sv.z = __expf(sv.z - m); sv.w = __expf(sv.w - m);
          sum += (sv.x + sv.y) + (sv.z + sv.w);
          *(float4*)(Sr + c) = sv;
        }
      } else {
        float4 z4 = {0.f, 0.f, 0.f, 0.f};
        for (int c = c0; c < c0 + T / 2; c += 4) *(float4*)(Sr + c) = z4;
      }
      sum += __shfl_xor(sum, 16, 32);
      float inv = 1.f / sum;
      for (int c = c0; c < c0 + T / 2; c += 4) {
        float4 sv = *(const float4*)(Sr + c);
        v4bf o;
        o[0] = (__bf16)(sv.x * inv); o[1] = (__bf16)(sv.y * inv);
        o[2] = (__bf16)(sv.z * inv); o[3] = (__bf16)(sv.w * inv);
        *(v4bf*)(Sbr + row * TK + c) = o;
      }
      if (WS == 2) {                                    // pad attn cols 16..31
        v4bf zb = {};
        *(v4bf*)(Sbr + row * TK + 16 + lh * 8) = zb;
        *(v4bf*)(Sbr + row * TK + 20 + lh * 8) = zb;
      }
    }
    // Y_i = softmax(S) * V   (K-dim = tokens, N = 32 channels = 2 n-tiles)
#pragma unroll
    for (int nt = 0; nt < 2; ++nt) {
      v8f acc = {};
      for (int kk = 0; kk < NKT; ++kk) {
        v16bf as = *(const v16bf*)(Sbr + ln * TK + kk * 32 + lh * 16);
        v16bf bv = *(const v16bf*)(Vs + (nt * 16 + ln) * TK + kk * 32 + lh * 16);
        acc = WMMA_BF16(as, bv, acc);
      }
#pragma unroll
      for (int r = 0; r < 8; ++r) {
        int i = it * 16 + r + 8 * lh;
        int vi = vox(i);                                // un-shift == same mapping
        int chan = S * 64 + head * 32 + nt * 16 + ln;
        Y[((size_t)b * VOX + vi) * CH + chan] = (__bf16)acc[r];
      }
    }
  }
}

// ---------------------------------------------------------------------------
// Kernel 3: output projection 192->192, f32 output in [b][c][v] order.
// Per-wave fixed n-tile -> the 6 B-operands (Y columns) are loaded once.
// ---------------------------------------------------------------------------
__global__ __launch_bounds__(256) void k_proj_out(
    const __bf16* __restrict__ Y, const __bf16* __restrict__ WoutB,
    const float* __restrict__ Bout, float* __restrict__ O)
{
  const int tid = threadIdx.x;
  const int gv0 = blockIdx.x * 64;
  const int b   = gv0 >> 15;
  const int v0  = gv0 & (VOX - 1);
  const int wave = tid >> 5, lane = tid & 31, ln = lane & 15, lh = lane >> 4;
  const int n0 = (wave & 3) * 16;                       // fixed per wave

  v16bf bvs[6];
#pragma unroll
  for (int kk = 0; kk < 6; ++kk)
    bvs[kk] = *(const v16bf*)(Y + ((size_t)(b * VOX) + v0 + n0 + ln) * CH +
                              kk * 32 + lh * 16);

  for (int tile = wave; tile < 48; tile += 8) {         // m0 varies, n0 fixed
    const int m0 = (tile >> 2) * 16;
    v8f acc = {};
#pragma unroll
    for (int kk = 0; kk < 6; ++kk) {
      v16bf a = *(const v16bf*)(WoutB + (size_t)(m0 + ln) * CH + kk * 32 + lh * 16);
      acc = WMMA_BF16(a, bvs[kk], acc);
    }
#pragma unroll
    for (int r = 0; r < 8; ++r) {
      int o = m0 + r + 8 * lh;
      O[(size_t)(b * CH + o) * VOX + v0 + n0 + ln] = acc[r] + Bout[o];
    }
  }
}

// ---------------------------------------------------------------------------
extern "C" void kernel_launch(void* const* d_in, const int* in_sizes, int n_in,
                              void* d_out, int out_size, void* d_ws, size_t ws_size,
                              hipStream_t stream)
{
  const float* x     = (const float*)d_in[0];
  const float* ln_g  = (const float*)d_in[1];
  const float* ln_b  = (const float*)d_in[2];
  const float* w_in  = (const float*)d_in[3];
  const float* b_in  = (const float*)d_in[4];
  const float* w_out = (const float*)d_in[5];
  const float* b_out = (const float*)d_in[6];
  float* out = (float*)d_out;

  const size_t P_BYTES = (size_t)NB * 3 * CH * VOX * sizeof(__bf16); // 75.5 MB
  const size_t Y_BYTES = (size_t)NB * VOX * CH * sizeof(__bf16);     // 25.2 MB
  __bf16* P     = (__bf16*)d_ws;
  __bf16* Yb    = (__bf16*)((char*)d_ws + P_BYTES);
  __bf16* WinB  = (__bf16*)((char*)d_ws + P_BYTES + Y_BYTES);
  __bf16* WoutB = WinB + W_IN_ELEMS;

  k_cvt_weights<<<(W_IN_ELEMS + 255) / 256, 256, 0, stream>>>(w_in, w_out, WinB, WoutB);
  k_ln_proj<<<1024, 256, 0, stream>>>(x, ln_g, ln_b, WinB, b_in, P);
  k_attn<2><<<NB * 2048 * 2, 128, 0, stream>>>(P, Yb);
  k_attn<4><<<NB * 512 * 2, 128, 0, stream>>>(P, Yb);
  k_attn<8><<<NB * 64 * 2, 128, 0, stream>>>(P, Yb);
  k_proj_out<<<1024, 256, 0, stream>>>(Yb, WoutB, b_out, out);
}
